// CodeShellAttention_47519518163076
// MI455X (gfx1250) — matrix operand order repaired and checked by reading, compiled, not run on hardware
//
#include <hip/hip_runtime.h>
#include <stdint.h>

#define HIDDEN 4096
#define NHEADS 32
#define NKV    8
#define HD     128
#define KVDIM  1024
#define SEQ    2048
#define QKVN   6144   // HIDDEN + 2*KVDIM

typedef __bf16 bf16_t;
typedef __attribute__((ext_vector_type(16))) __bf16 v16bf;
typedef __attribute__((ext_vector_type(8)))  float  v8f;
typedef __attribute__((ext_vector_type(4)))  unsigned int v4u;

struct B32 { v4u lo, hi; };

// fp32 -> bf16 round-to-nearest-even (bit twiddle; no reliance on target cvt)
__device__ __forceinline__ bf16_t f2bf(float f) {
  unsigned int u = __builtin_bit_cast(unsigned int, f);
  unsigned int r = (u + 0x7FFFu + ((u >> 16) & 1u)) >> 16;
  unsigned short h = (unsigned short)r;
  return __builtin_bit_cast(bf16_t, h);
}

// A-fragment (16x32 bf16): two 16B chunks per lane (K ks..ks+7, ks+8..ks+15)
__device__ __forceinline__ v16bf load_frag_2x16(const bf16_t* p0, const bf16_t* p1) {
  B32 t;
  t.lo = *reinterpret_cast<const v4u*>(p0);
  t.hi = *reinterpret_cast<const v4u*>(p1);
  return __builtin_bit_cast(v16bf, t);
}
// B-fragment (32x16 bf16): one contiguous 32B chunk per lane (16 K values)
__device__ __forceinline__ v16bf load_frag_32B(const bf16_t* p) {
  B32 t;
  t.lo = *reinterpret_cast<const v4u*>(p);
  t.hi = *reinterpret_cast<const v4u*>(p + 8);
  return __builtin_bit_cast(v16bf, t);
}

// CDNA5 async direct global->LDS copy, 16B per lane, tracked by ASYNCcnt.
// LDS destination = low 32 bits of the generic pointer (LDS aperture rule).
__device__ __forceinline__ void async_cp16(void* lds_dst, const void* gsrc) {
  uint32_t lo = (uint32_t)(uintptr_t)lds_dst;
  uint64_t ga = (uint64_t)(uintptr_t)gsrc;
  asm volatile("global_load_async_to_lds_b128 %0, %1, off"
               :: "v"(lo), "v"(ga) : "memory");
}

// ---------------------------------------------------------------------------
// fp32 -> bf16 flat convert
// ---------------------------------------------------------------------------
__global__ void convert_f32_bf16(const float* __restrict__ x,
                                 bf16_t* __restrict__ y, int n) {
  int i = blockIdx.x * blockDim.x + threadIdx.x;
  if (i < n) y[i] = f2bf(x[i]);
}

// ---------------------------------------------------------------------------
// W[K][N] f32  ->  Wt[N][K] bf16   (LDS-tiled transpose)
// ---------------------------------------------------------------------------
__global__ __launch_bounds__(256) void transpose_to_bf16(
    const float* __restrict__ W, bf16_t* __restrict__ Wt, int K, int N) {
  __shared__ float tile[32][33];
  const int n0 = blockIdx.x * 32, k0 = blockIdx.y * 32;
  const int tx = threadIdx.x, ty = threadIdx.y;   // 32 x 8
  for (int j = 0; j < 32; j += 8)
    tile[ty + j][tx] = W[(size_t)(k0 + ty + j) * N + n0 + tx];
  __syncthreads();
  for (int j = 0; j < 32; j += 8)
    Wt[(size_t)(n0 + ty + j) * K + k0 + tx] = f2bf(tile[tx][ty + j]);
}

// ---------------------------------------------------------------------------
// C[M][N] f32 = A[M][K]bf16 @ Bt[N][K]bf16^T + bias
// 128x128 block tile, 8 waves (4Mx2N), each wave 32x64 via 2x4 wmma frags.
// Double-buffered LDS fed by async global->LDS copies (ASYNCcnt pipelined).
// ---------------------------------------------------------------------------
__global__ __launch_bounds__(256) void wmma_gemm_bf16(
    const bf16_t* __restrict__ A, const bf16_t* __restrict__ Bt,
    const float* __restrict__ bias, float* __restrict__ C,
    int M, int N, int K) {
  __shared__ bf16_t As[2][128][48];   // 96B rows -> 16B aligned chunks
  __shared__ bf16_t Bs[2][128][48];
  const int tid  = threadIdx.x;
  const int lane = tid & 31, wave = tid >> 5;
  const int wm = wave >> 1, wn = wave & 1;
  const int m0 = blockIdx.y * 128, n0 = blockIdx.x * 128;
  const int lrow = lane & 15, lhalf = lane >> 4;

  // This thread's two staging chunks (16B each per matrix per tile)
  const int c0 = tid,        r0 = c0 >> 2, o0 = (c0 & 3) * 8;
  const int c1 = tid + 256,  r1 = c1 >> 2, o1 = (c1 & 3) * 8;

  v8f zero = {};
  v8f acc[2][4];
  for (int i = 0; i < 2; ++i)
    for (int j = 0; j < 4; ++j) acc[i][j] = zero;

  const int TK = K >> 5;     // number of 32-wide k tiles

  // stage tile t into buffer b: 4 async copies per thread (2 A + 2 B)
  auto stage = [&](int t, int b) {
    const int k0 = t << 5;
    async_cp16(&As[b][r0][o0], &A [(size_t)(m0 + r0) * K + k0 + o0]);
    async_cp16(&Bs[b][r0][o0], &Bt[(size_t)(n0 + r0) * K + k0 + o0]);
    async_cp16(&As[b][r1][o1], &A [(size_t)(m0 + r1) * K + k0 + o1]);
    async_cp16(&Bs[b][r1][o1], &Bt[(size_t)(n0 + r1) * K + k0 + o1]);
  };

  stage(0, 0);
  for (int t = 0; t < TK; ++t) {
    const int buf = t & 1;
    if (t + 1 < TK) {
      stage(t + 1, buf ^ 1);
      // 8 outstanding per wave; <=4 means tile t's copies landed (in-order)
      asm volatile("s_wait_asynccnt 0x4" ::: "memory");
    } else {
      asm volatile("s_wait_asynccnt 0x0" ::: "memory");
    }
    __syncthreads();   // all waves' copies for tile t visible in LDS

    v16bf af[2], bfr[4];
    for (int mt = 0; mt < 2; ++mt) {
      int r = wm * 32 + mt * 16 + lrow;
      int ks = lhalf * 16;
      af[mt] = load_frag_2x16(&As[buf][r][ks], &As[buf][r][ks + 8]);
    }
    for (int nt = 0; nt < 4; ++nt) {
      int cc = wn * 64 + nt * 16 + lrow;
      bfr[nt] = load_frag_32B(&Bs[buf][cc][lhalf * 16]);
    }
    for (int mt = 0; mt < 2; ++mt)
      for (int nt = 0; nt < 4; ++nt)
        acc[mt][nt] = __builtin_amdgcn_wmma_f32_16x16x32_bf16(
            false, af[mt], false, bfr[nt], (short)0, acc[mt][nt], false, false);
    __syncthreads();   // done reading buf; next iter may overwrite it
  }

  for (int mt = 0; mt < 2; ++mt)
    for (int nt = 0; nt < 4; ++nt) {
      int col = n0 + wn * 64 + nt * 16 + lrow;
      float b = bias[col];
      for (int g = 0; g < 8; ++g) {
        int row = m0 + wm * 32 + mt * 16 + g + 8 * lhalf;
        C[(size_t)row * N + col] = acc[mt][nt][g] + b;
      }
    }
}

// ---------------------------------------------------------------------------
// RoPE + repack: qkv f32 [SEQ][QKVN] ->
//   Qb bf16 [SEQ][HIDDEN]            (row-major: A-frag friendly)
//   Kb bf16 [NKV][SEQ][HD]           (key-major: B-frag friendly for QK^T)
//   Vt bf16 [NKV][HD][SEQ]           (d-major:   B-frag friendly for P@V)
// ---------------------------------------------------------------------------
__global__ __launch_bounds__(256) void rope_pack(
    const float* __restrict__ qkv, const int* __restrict__ pos,
    bf16_t* __restrict__ Qb, bf16_t* __restrict__ Kb, bf16_t* __restrict__ Vt) {
  const int s = blockIdx.x;
  const int t = threadIdx.x;
  const float p = (float)pos[s];
  const float c_ln = 0.14391156831f;       // ln(10000)/64
  const size_t base = (size_t)s * QKVN;

  for (int i = t; i < NHEADS * 64; i += 256) {   // Q: 32 heads x 64 pairs
    int hh = i >> 6, j = i & 63;
    float ang = p * __expf(-c_ln * (float)j);
    float sn, cs; __sincosf(ang, &sn, &cs);
    float x1 = qkv[base + hh * HD + j];
    float x2 = qkv[base + hh * HD + j + 64];
    size_t ob = (size_t)s * HIDDEN + hh * HD;
    Qb[ob + j]      = f2bf(x1 * cs - x2 * sn);
    Qb[ob + j + 64] = f2bf(x2 * cs + x1 * sn);
  }
  for (int i = t; i < NKV * 64; i += 256) {      // K: 8 heads x 64 pairs
    int hh = i >> 6, j = i & 63;
    float ang = p * __expf(-c_ln * (float)j);
    float sn, cs; __sincosf(ang, &sn, &cs);
    float x1 = qkv[base + HIDDEN + hh * HD + j];
    float x2 = qkv[base + HIDDEN + hh * HD + j + 64];
    size_t ob = ((size_t)hh * SEQ + s) * HD;
    Kb[ob + j]      = f2bf(x1 * cs - x2 * sn);
    Kb[ob + j + 64] = f2bf(x2 * cs + x1 * sn);
  }
  for (int i = t; i < KVDIM; i += 256) {         // V: transpose to d-major
    int hh = i >> 7, d = i & 127;
    Vt[((size_t)hh * HD + d) * SEQ + s] =
        f2bf(qkv[base + HIDDEN + KVDIM + hh * HD + d]);
  }
}

// ---------------------------------------------------------------------------
// Flash attention: grid (SEQ/64 q-tiles, 32 heads), 4 waves x 16 q-rows.
// S = Q@K^T via 4 wmma/score-frag; online softmax via 16-lane shfl_xor;
// P routed through per-wave LDS (C-layout -> A-layout); O += P@V via wmma.
// ---------------------------------------------------------------------------
__global__ __launch_bounds__(128) void flash_attn(
    const bf16_t* __restrict__ Qb, const bf16_t* __restrict__ Kb,
    const bf16_t* __restrict__ Vt, bf16_t* __restrict__ Ob) {
  __shared__ bf16_t Ps[4][16][48];
  const int h  = blockIdx.y, hk = h >> 2;        // GQA: 4 q-heads per kv-head
  const int q0 = blockIdx.x * 64;
  const int lane = threadIdx.x & 31, wave = threadIdx.x >> 5;
  const int lrow = lane & 15, lhalf = lane >> 4;
  const int qrow0 = q0 + wave * 16;
  const float scale = 0.0883883476483184f;       // 1/sqrt(128)

  // Q A-fragments: 4 d-chunks of 32, resident for whole kv loop
  v16bf qf[4];
  {
    const bf16_t* qb = Qb + (size_t)(qrow0 + lrow) * HIDDEN + h * HD;
    int ks = lhalf * 16;
    for (int dc = 0; dc < 4; ++dc)
      qf[dc] = load_frag_2x16(qb + dc * 32 + ks, qb + dc * 32 + ks + 8);
  }

  v8f zero = {};
  v8f o[8];
  for (int dt = 0; dt < 8; ++dt) o[dt] = zero;
  float mrow[8], lsum[8];
  for (int g = 0; g < 8; ++g) { mrow[g] = -3.0e38f; lsum[g] = 0.f; }

  for (int kv0 = 0; kv0 < q0 + 64; kv0 += 32) {
    // ---- scores: S[16 x 32] = Q @ K^T -------------------------------------
    v8f sfr[2];
    for (int nt = 0; nt < 2; ++nt) {
      v8f s = zero;
      int kvcol = kv0 + nt * 16 + lrow;
      const bf16_t* kb = Kb + ((size_t)hk * SEQ + kvcol) * HD;
      for (int dc = 0; dc < 4; ++dc) {
        v16bf kf = load_frag_32B(kb + dc * 32 + lhalf * 16);
        s = __builtin_amdgcn_wmma_f32_16x16x32_bf16(
            false, qf[dc], false, kf, (short)0, s, false, false);
      }
      sfr[nt] = s;
    }
    // ---- scale + causal mask ---------------------------------------------
    for (int nt = 0; nt < 2; ++nt) {
      int col = kv0 + nt * 16 + lrow;
      for (int g = 0; g < 8; ++g) {
        int row = qrow0 + g + 8 * lhalf;
        float v = sfr[nt][g] * scale;
        sfr[nt][g] = (col <= row) ? v : -3.0e38f;
      }
    }
    // ---- online softmax (rows live across 16-lane half-groups) -----------
    float tm[8];
    for (int g = 0; g < 8; ++g) tm[g] = fmaxf(sfr[0][g], sfr[1][g]);
    for (int off = 1; off < 16; off <<= 1)
      for (int g = 0; g < 8; ++g)
        tm[g] = fmaxf(tm[g], __shfl_xor(tm[g], off, 32));
    float alpha[8];
    for (int g = 0; g < 8; ++g) {
      float mn = fmaxf(mrow[g], tm[g]);
      alpha[g] = __expf(mrow[g] - mn);
      mrow[g]  = mn;
    }
    float rs[8];
    for (int g = 0; g < 8; ++g) rs[g] = 0.f;
    for (int nt = 0; nt < 2; ++nt)
      for (int g = 0; g < 8; ++g) {
        float pv = __expf(sfr[nt][g] - mrow[g]);
        sfr[nt][g] = pv;
        rs[g] += pv;
      }
    for (int off = 1; off < 16; off <<= 1)
      for (int g = 0; g < 8; ++g) rs[g] += __shfl_xor(rs[g], off, 32);
    for (int g = 0; g < 8; ++g) lsum[g] = lsum[g] * alpha[g] + rs[g];
    for (int dt = 0; dt < 8; ++dt)
      for (int g = 0; g < 8; ++g) o[dt][g] = o[dt][g] * alpha[g];

    // ---- P: C-layout -> bf16 A-layout via per-wave LDS --------------------
    for (int nt = 0; nt < 2; ++nt)
      for (int g = 0; g < 8; ++g)
        Ps[wave][g + 8 * lhalf][nt * 16 + lrow] = f2bf(sfr[nt][g]);
    asm volatile("s_wait_dscnt 0x0" ::: "memory");
    v16bf pf = load_frag_2x16(&Ps[wave][lrow][lhalf * 16],
                              &Ps[wave][lrow][lhalf * 16 + 8]);

    // ---- O += P @ V (V stored d-major -> contiguous B frags) --------------
    for (int dt = 0; dt < 8; ++dt) {
      const bf16_t* vb =
          Vt + ((size_t)hk * HD + dt * 16 + lrow) * SEQ + kv0 + lhalf * 16;
      v16bf vf = load_frag_32B(vb);
      o[dt] = __builtin_amdgcn_wmma_f32_16x16x32_bf16(
          false, pf, false, vf, (short)0, o[dt], false, false);
    }
  }

  // normalize + store bf16 (feeds final GEMM)
  for (int dt = 0; dt < 8; ++dt) {
    int col = h * HD + dt * 16 + lrow;
    for (int g = 0; g < 8; ++g) {
      int row = qrow0 + g + 8 * lhalf;
      Ob[(size_t)row * HIDDEN + col] = f2bf(o[dt][g] / lsum[g]);
    }
  }
}

// ---------------------------------------------------------------------------
extern "C" void kernel_launch(void* const* d_in, const int* in_sizes, int n_in,
                              void* d_out, int out_size, void* d_ws,
                              size_t ws_size, hipStream_t stream) {
  const int*   positions = (const int*)d_in[0];
  const float* X         = (const float*)d_in[1];
  const float* Wqkv      = (const float*)d_in[2];
  const float* bqkv      = (const float*)d_in[3];
  const float* Wproj     = (const float*)d_in[4];
  const float* bproj     = (const float*)d_in[5];
  float* out = (float*)d_out;

  char* p = (char*)d_ws;
  bf16_t* Xb     = (bf16_t*)p; p += (size_t)SEQ * HIDDEN * 2;
  bf16_t* WqkvT  = (bf16_t*)p; p += (size_t)QKVN * HIDDEN * 2;
  bf16_t* WprojT = (bf16_t*)p; p += (size_t)HIDDEN * HIDDEN * 2;
  float*  qkv    = (float*) p; p += (size_t)SEQ * QKVN * 4;
  bf16_t* Qb     = (bf16_t*)p; p += (size_t)SEQ * HIDDEN * 2;
  bf16_t* Kb     = (bf16_t*)p; p += (size_t)NKV * SEQ * HD * 2;
  bf16_t* Vt     = (bf16_t*)p; p += (size_t)NKV * HD * SEQ * 2;
  bf16_t* attnB  = (bf16_t*)p; p += (size_t)SEQ * HIDDEN * 2;

  convert_f32_bf16<<<(SEQ * HIDDEN + 255) / 256, 256, 0, stream>>>(
      X, Xb, SEQ * HIDDEN);
  transpose_to_bf16<<<dim3(QKVN / 32, HIDDEN / 32), dim3(32, 8), 0, stream>>>(
      Wqkv, WqkvT, HIDDEN, QKVN);
  transpose_to_bf16<<<dim3(HIDDEN / 32, HIDDEN / 32), dim3(32, 8), 0, stream>>>(
      Wproj, WprojT, HIDDEN, HIDDEN);
  wmma_gemm_bf16<<<dim3(QKVN / 128, SEQ / 128), 256, 0, stream>>>(
      Xb, WqkvT, bqkv, qkv, SEQ, QKVN, HIDDEN);
  rope_pack<<<SEQ, 256, 0, stream>>>(qkv, positions, Qb, Kb, Vt);
  flash_attn<<<dim3(SEQ / 64, NHEADS), 128, 0, stream>>>(Qb, Kb, Vt, attnB);
  wmma_gemm_bf16<<<dim3(HIDDEN / 128, SEQ / 128), 256, 0, stream>>>(
      attnB, WprojT, bproj, out, SEQ, HIDDEN, HIDDEN);
}
